// MultiModalModel_26766236188705
// MI455X (gfx1250) — matrix operand order repaired and checked by reading, not compiled, not run on hardware
//
#include <hip/hip_runtime.h>

// ---------------------------------------------------------------------------
// Model constants
// ---------------------------------------------------------------------------
#define B_ 32
#define K_ 32
#define NN_ 2048
#define EE_ 16384
#define DD_ 768
#define L_GNN_ 2
#define L_TR_ 4
#define HH_ 12
#define DH_ 64
#define FF_ 3072
#define BK_D_ (B_ * K_ * DD_)   // 786432

typedef __attribute__((ext_vector_type(16))) __bf16 v16bf;
typedef __attribute__((ext_vector_type(8)))  float  v8f;
typedef __attribute__((ext_vector_type(4)))  unsigned int u32x4;
typedef __attribute__((ext_vector_type(8)))  int i32x8;
typedef __attribute__((ext_vector_type(4)))  int i32x4;

#if __has_builtin(__builtin_amdgcn_tensor_load_to_lds) && \
    __has_builtin(__builtin_amdgcn_s_wait_tensorcnt)
#define USE_TDM 1
#else
#define USE_TDM 0
#endif

static __device__ __forceinline__ unsigned short f2bf(float f) {
  union { float f; unsigned int u; } x; x.f = f;
  unsigned int u = x.u;
  return (unsigned short)((u + 0x7FFFu + ((u >> 16) & 1u)) >> 16);
}

static __device__ __forceinline__ float gelu_tanh(float x) {
  float x3 = x * x * x;
  return 0.5f * x * (1.f + tanhf(0.7978845608f * (x + 0.044715f * x3)));
}

#if USE_TDM
// Issue a TDM load of a [128 rows x 32 cols] bf16 tile (row stride Kd elems)
// from global tile-start `gbase` into LDS byte offset `lds_off`.
// D# packing per CDNA5 ISA (2D tensor, groups 2/3 zero).
static __device__ __forceinline__ void tdm_load_tile(const unsigned short* gbase,
                                                     unsigned lds_off, int Kd) {
  unsigned long long ga = (unsigned long long)(size_t)gbase;
  u32x4 g0;
  g0.x = 1u;                                                  // count=1, user mode
  g0.y = lds_off;                                             // lds_addr (bytes)
  g0.z = (unsigned)(ga & 0xFFFFFFFFull);                      // global_addr[31:0]
  g0.w = (unsigned)((ga >> 32) & 0x01FFFFFFull) | 0x80000000u; // addr[56:32] | type=2
  i32x8 g1;
  g1[0] = 1 << 16;                           // workgroup_mask=0, data_size=1 (2 bytes)
  g1[1] = (Kd & 0xFFFF) << 16;               // tensor_dim0[15:0]  (bits 63:48)
  g1[2] = ((Kd >> 16) & 0xFFFF) | (128 << 16); // tensor_dim0[31:16] | tensor_dim1[15:0]=128
  g1[3] = (32 << 16);                        // tensor_dim1[31:16]=0 | tile_dim0=32
  g1[4] = 128;                               // tile_dim1=128, tile_dim2=0
  g1[5] = Kd;                                // tensor_dim0_stride[31:0]
  g1[6] = 0;                                 // stride0[47:32]=0, stride1[15:0]=0
  g1[7] = 0;
  i32x4 z4 = {0, 0, 0, 0};                   // groups 2/3: dims/tiles unused (2D)
#if defined(__clang_major__) && __clang_major__ >= 23
  i32x8 z8 = {0, 0, 0, 0, 0, 0, 0, 0};
  __builtin_amdgcn_tensor_load_to_lds(g0, g1, z4, z4, z8, 0);
#else
  __builtin_amdgcn_tensor_load_to_lds(g0, g1, z4, z4, 0);
#endif
}
#endif

// ---------------------------------------------------------------------------
// Elementwise / utility kernels
// ---------------------------------------------------------------------------
__global__ __launch_bounds__(256) void zero_kernel(float* p, int n) {
  int i = blockIdx.x * 256 + threadIdx.x;
  if (i < n) p[i] = 0.f;
}

__global__ __launch_bounds__(256) void conv_bf16_kernel(const float* __restrict__ in,
                                                        unsigned short* __restrict__ out,
                                                        int n) {
  int i = blockIdx.x * 256 + threadIdx.x;
  if (i < n) out[i] = f2bf(in[i]);
}

// weight convert + transpose: in[k*Nd + n] (row-major KxN) -> out[n*Kd + k]
__global__ __launch_bounds__(256) void convT_bf16_kernel(const float* __restrict__ in,
                                                         unsigned short* __restrict__ out,
                                                         int Nd, int Kd) {
  int i = blockIdx.x * 256 + threadIdx.x;
  if (i >= Nd * Kd) return;
  int n = i / Kd, k = i - n * Kd;
  out[i] = f2bf(in[(size_t)k * Nd + n]);
}

__global__ __launch_bounds__(256) void gather_kernel(const float* __restrict__ emb,
                                                     const int* __restrict__ ids,
                                                     float* __restrict__ h) {
  int i = blockIdx.x * 256 + threadIdx.x;      // over N*D
  int n = i / DD_;
  int d = i - n * DD_;
  h[i] = emb[(size_t)ids[n] * DD_ + d];
}

__global__ __launch_bounds__(256) void scatter_add_kernel(const float* __restrict__ h,
                                                          const int* __restrict__ src,
                                                          const int* __restrict__ dst,
                                                          float* __restrict__ agg) {
  int i = blockIdx.x * 256 + threadIdx.x;      // over E*D
  int e = i / DD_;
  int d = i - e * DD_;
  atomicAdd(&agg[(size_t)dst[e] * DD_ + d], h[(size_t)src[e] * DD_ + d]);
}

__global__ __launch_bounds__(256) void add_conv_kernel(const float* __restrict__ a,
                                                       const float* __restrict__ b,
                                                       unsigned short* __restrict__ out,
                                                       int n) {
  int i = blockIdx.x * 256 + threadIdx.x;
  if (i < n) out[i] = f2bf(a[i] + b[i]);
}

__global__ __launch_bounds__(256) void readout_kernel(const float* __restrict__ h,
                                                      const int* __restrict__ batch,
                                                      float* __restrict__ sums,
                                                      float* __restrict__ cnt) {
  int i = blockIdx.x * 256 + threadIdx.x;      // over N*D
  int n = i / DD_;
  int d = i - n * DD_;
  int b = batch[n];
  atomicAdd(&sums[(size_t)b * DD_ + d], h[i]);
  if (d == 0) atomicAdd(&cnt[b], 1.f);
}

__global__ __launch_bounds__(256) void pool_accum_kernel(const float* __restrict__ sums,
                                                         const float* __restrict__ cnt,
                                                         float* __restrict__ pool_k) {
  int i = blockIdx.x * 256 + threadIdx.x;      // over B*D
  if (i >= B_ * DD_) return;
  int b = i / DD_;
  pool_k[i] += sums[i] / fmaxf(cnt[b], 1.f);
}

__global__ __launch_bounds__(256) void build_feats_kernel(
    const float* __restrict__ raw, const float* __restrict__ pool,
    const int* __restrict__ role_ids, const int* __restrict__ pos_ids,
    const int* __restrict__ hop_ids, const float* __restrict__ role_emb,
    const float* __restrict__ pos_emb, const float* __restrict__ hop_emb,
    float* __restrict__ out) {
  int i = blockIdx.x * 256 + threadIdx.x;      // over B*K*D
  if (i >= BK_D_) return;
  int d = i % DD_;
  int bk = i / DD_;
  int k = bk % K_;
  int b = bk / K_;
  float v = raw[i] + pool[((size_t)k * B_ + b) * DD_ + d];
  v += role_emb[(size_t)role_ids[bk] * DD_ + d];
  v += pos_emb[(size_t)pos_ids[bk] * DD_ + d];
  v += hop_emb[(size_t)hop_ids[bk] * DD_ + d];
  out[i] = v;
}

// LayerNorm over last dim (D=768); one block per row; optional residual add.
__global__ __launch_bounds__(256) void ln_kernel(const float* __restrict__ x,
                                                 const float* __restrict__ res,
                                                 const float* __restrict__ s,
                                                 const float* __restrict__ b,
                                                 float* __restrict__ out) {
  int row = blockIdx.x;
  int tid = threadIdx.x;
  __shared__ float buf[DD_];
  __shared__ float red[256];
  const size_t base = (size_t)row * DD_;
  float lsum = 0.f;
  for (int i = tid; i < DD_; i += 256) {
    float v = x[base + i];
    if (res) v += res[base + i];
    buf[i] = v;
    lsum += v;
  }
  red[tid] = lsum; __syncthreads();
  for (int st = 128; st > 0; st >>= 1) { if (tid < st) red[tid] += red[tid + st]; __syncthreads(); }
  float mean = red[0] * (1.f / DD_);
  __syncthreads();
  float lvar = 0.f;
  for (int i = tid; i < DD_; i += 256) { float d = buf[i] - mean; lvar += d * d; }
  red[tid] = lvar; __syncthreads();
  for (int st = 128; st > 0; st >>= 1) { if (tid < st) red[tid] += red[tid + st]; __syncthreads(); }
  float inv = rsqrtf(red[0] * (1.f / DD_) + 1e-12f);
  for (int i = tid; i < DD_; i += 256)
    out[base + i] = (buf[i] - mean) * inv * s[i] + b[i];
}

// Attention for one (batch, head): q,k,v are [B*K, D] with head slice h*64.
__global__ __launch_bounds__(256) void attn_kernel(const float* __restrict__ q,
                                                   const float* __restrict__ k,
                                                   const float* __restrict__ v,
                                                   float* __restrict__ ctx) {
  int b = blockIdx.x / HH_;
  int h = blockIdx.x % HH_;
  __shared__ float sq[K_][DH_];
  __shared__ float sk[K_][DH_];
  __shared__ float sv[K_][DH_];
  __shared__ float ss[K_][K_];
  for (int i = threadIdx.x; i < K_ * DH_; i += 256) {
    int n = i >> 6, d = i & 63;
    size_t off = ((size_t)b * K_ + n) * DD_ + h * DH_ + d;
    sq[n][d] = q[off]; sk[n][d] = k[off]; sv[n][d] = v[off];
  }
  __syncthreads();
  for (int i = threadIdx.x; i < K_ * K_; i += 256) {
    int r = i >> 5, c = i & 31;
    float s = 0.f;
#pragma unroll 8
    for (int d = 0; d < DH_; ++d) s += sq[r][d] * sk[c][d];
    ss[r][c] = s * 0.125f;   // 1/sqrt(64)
  }
  __syncthreads();
  if (threadIdx.x < K_) {
    int r = threadIdx.x;
    float mx = -1e30f;
    for (int c = 0; c < K_; ++c) mx = fmaxf(mx, ss[r][c]);
    float sum = 0.f;
    for (int c = 0; c < K_; ++c) { float e = __expf(ss[r][c] - mx); ss[r][c] = e; sum += e; }
    float inv = 1.f / sum;
    for (int c = 0; c < K_; ++c) ss[r][c] *= inv;
  }
  __syncthreads();
  for (int i = threadIdx.x; i < K_ * DH_; i += 256) {
    int n = i >> 6, d = i & 63;
    float s = 0.f;
#pragma unroll 8
    for (int j = 0; j < K_; ++j) s += ss[n][j] * sv[j][d];
    ctx[((size_t)b * K_ + n) * DD_ + h * DH_ + d] = s;
  }
}

// ---------------------------------------------------------------------------
// WMMA bf16 GEMM: C[M,Nd] = act(A[M,Kd] @ W[Kd,Nd] + bias), W given as Wt[Nd][Kd]
// Block: 256 threads (8 waves), 128x128 macro tile, BK=32.
// Wave (wm 0..3, wn 0..1) computes a 32x64 sub-tile = 2x4 WMMA 16x16 tiles.
// Tile staging: TDM tensor_load_to_lds when available, else manual b128 copies.
// ---------------------------------------------------------------------------
union FragBF { v16bf v; u32x4 q[2]; };

__global__ __launch_bounds__(256)
void gemm_bf16_kernel(const unsigned short* __restrict__ A,
                      const unsigned short* __restrict__ Bt,
                      const float* __restrict__ bias,
                      float* __restrict__ C,
                      int Nd, int Kd, int act) {
  extern __shared__ unsigned short smem[];
  unsigned short* sA = smem;            // LDS bytes [0, 8192)
  unsigned short* sB = smem + 4096;     // LDS bytes [8192, 16384)
  const int tid  = threadIdx.x;
  const int lane = tid & 31;
  const int wave = tid >> 5;
  const int wm = wave >> 1;
  const int wn = wave & 1;
  const int m0 = blockIdx.y * 128;
  const int n0 = blockIdx.x * 128;

  const v8f vzero = {0.f, 0.f, 0.f, 0.f, 0.f, 0.f, 0.f, 0.f};
  v8f acc[2][4];
#pragma unroll
  for (int mi = 0; mi < 2; ++mi)
#pragma unroll
    for (int ni = 0; ni < 4; ++ni) acc[mi][ni] = vzero;

  const int la  = lane & 15;
  const int kbA = (lane < 16) ? 0 : 8;    // A-frag K base (16-bit A layout)
  const int kbB = (lane < 16) ? 0 : 8;    // B-frag K base (16-bit table layout, matches A)

  for (int k0 = 0; k0 < Kd; k0 += 32) {
    __syncthreads();                      // previous-iteration LDS readers done
#if USE_TDM
    if (wave == 0) {
      tdm_load_tile(A + (size_t)m0 * Kd + k0, 0u, Kd);
      tdm_load_tile(Bt + (size_t)n0 * Kd + k0, 8192u, Kd);
      __builtin_amdgcn_s_wait_tensorcnt(0);
    }
#else
    for (int i = tid; i < 512; i += 256) {
      int r = i >> 2;
      int c = (i & 3) << 3;               // 8 bf16 = 16 bytes
      *(u32x4*)(&sA[r * 32 + c]) = *(const u32x4*)(A + (size_t)(m0 + r) * Kd + k0 + c);
      *(u32x4*)(&sB[r * 32 + c]) = *(const u32x4*)(Bt + (size_t)(n0 + r) * Kd + k0 + c);
      if (k0 + 32 < Kd)
        __builtin_prefetch(A + (size_t)(m0 + r) * Kd + k0 + 32 + c, 0, 0);
    }
#endif
    __syncthreads();

    FragBF a[2], b[4];
#pragma unroll
    for (int mi = 0; mi < 2; ++mi) {
      int base = (wm * 32 + mi * 16 + la) * 32;
      a[mi].q[0] = *(const u32x4*)(&sA[base + kbA]);
      a[mi].q[1] = *(const u32x4*)(&sA[base + kbA + 16]);
    }
#pragma unroll
    for (int ni = 0; ni < 4; ++ni) {
      int base = (wn * 64 + ni * 16 + la) * 32;
      b[ni].q[0] = *(const u32x4*)(&sB[base + kbB]);
      b[ni].q[1] = *(const u32x4*)(&sB[base + kbB + 16]);
    }
#pragma unroll
    for (int mi = 0; mi < 2; ++mi)
#pragma unroll
      for (int ni = 0; ni < 4; ++ni)
        acc[mi][ni] = __builtin_amdgcn_wmma_f32_16x16x32_bf16(
            false, a[mi].v, false, b[ni].v, (short)0, acc[mi][ni], false, false);
  }

  const int rbase = (lane >> 4) << 3;     // 0 or 8
#pragma unroll
  for (int mi = 0; mi < 2; ++mi) {
#pragma unroll
    for (int ni = 0; ni < 4; ++ni) {
      int gn = n0 + wn * 64 + ni * 16 + la;
      float bv = bias ? bias[gn] : 0.f;
#pragma unroll
      for (int r = 0; r < 8; ++r) {
        int gm = m0 + wm * 32 + mi * 16 + rbase + r;
        float vv = acc[mi][ni][r] + bv;
        if (act == 1) vv = fmaxf(vv, 0.f);
        else if (act == 2) vv = gelu_tanh(vv);
        C[(size_t)gm * Nd + gn] = vv;
      }
    }
  }
}

// ---------------------------------------------------------------------------
// Host orchestration
// ---------------------------------------------------------------------------
extern "C" void kernel_launch(void* const* d_in, const int* in_sizes, int n_in,
                              void* d_out, int out_size, void* d_ws, size_t ws_size,
                              hipStream_t stream) {
  (void)in_sizes; (void)n_in; (void)out_size; (void)ws_size;
  const float* raw      = (const float*)d_in[0];
  const int*   x0       = (const int*)d_in[1];
  const int*   edge0    = (const int*)d_in[2];
  const int*   batch0   = (const int*)d_in[3];
  const int*   x1       = (const int*)d_in[4];
  const int*   edge1    = (const int*)d_in[5];
  const int*   batch1   = (const int*)d_in[6];
  const int*   role_ids = (const int*)d_in[7];
  const int*   pos_ids  = (const int*)d_in[8];
  const int*   hop_ids  = (const int*)d_in[9];
  const float* amino    = (const float*)d_in[10];
  const float* gnn_W    = (const float*)d_in[11];
  const float* gnn_b    = (const float*)d_in[12];
  const float* role_emb = (const float*)d_in[13];
  const float* pos_emb  = (const float*)d_in[14];
  const float* hop_emb  = (const float*)d_in[15];
  const float* eln_s    = (const float*)d_in[16];
  const float* eln_b    = (const float*)d_in[17];
  const float* Wq = (const float*)d_in[18]; const float* bq = (const float*)d_in[19];
  const float* Wk = (const float*)d_in[20]; const float* bk = (const float*)d_in[21];
  const float* Wv = (const float*)d_in[22]; const float* bv = (const float*)d_in[23];
  const float* Wo = (const float*)d_in[24]; const float* bo = (const float*)d_in[25];
  const float* ln1s = (const float*)d_in[26]; const float* ln1b = (const float*)d_in[27];
  const float* W1 = (const float*)d_in[28]; const float* b1 = (const float*)d_in[29];
  const float* W2 = (const float*)d_in[30]; const float* b2 = (const float*)d_in[31];
  const float* ln2s = (const float*)d_in[32]; const float* ln2b = (const float*)d_in[33];
  float* outp = (float*)d_out;

  // ---- workspace carve-up -------------------------------------------------
  char* w = (char*)d_ws;
  auto alloc = [&](size_t bytes) -> char* {
    char* p = w;
    w += (bytes + 255) & ~(size_t)255;
    return p;
  };
  const size_t DxD = (size_t)DD_ * DD_;
  unsigned short* gnnWb = (unsigned short*)alloc(2 * DxD * 2);
  unsigned short* Wqb   = (unsigned short*)alloc(4 * DxD * 2);
  unsigned short* Wkb   = (unsigned short*)alloc(4 * DxD * 2);
  unsigned short* Wvb   = (unsigned short*)alloc(4 * DxD * 2);
  unsigned short* Wob   = (unsigned short*)alloc(4 * DxD * 2);
  unsigned short* W1b   = (unsigned short*)alloc(4 * (size_t)DD_ * FF_ * 2);
  unsigned short* W2b   = (unsigned short*)alloc(4 * (size_t)FF_ * DD_ * 2);
  float* hnodeA = (float*)alloc((size_t)NN_ * DD_ * 4);
  float* hnodeB = (float*)alloc((size_t)NN_ * DD_ * 4);
  float* agg    = (float*)alloc((size_t)NN_ * DD_ * 4);
  unsigned short* xbf = (unsigned short*)alloc((size_t)NN_ * DD_ * 2);
  float* sums   = (float*)alloc((size_t)B_ * DD_ * 4);
  float* cnt    = (float*)alloc((size_t)B_ * 4);
  float* pool   = (float*)alloc((size_t)K_ * B_ * DD_ * 4);
  float* hA     = (float*)alloc((size_t)BK_D_ * 4);
  float* hB     = (float*)alloc((size_t)BK_D_ * 4);
  unsigned short* hbf = (unsigned short*)alloc((size_t)BK_D_ * 2);
  float* qb     = (float*)alloc((size_t)BK_D_ * 4);
  float* kb2    = (float*)alloc((size_t)BK_D_ * 4);
  float* vb2    = (float*)alloc((size_t)BK_D_ * 4);
  float* ctx    = (float*)alloc((size_t)BK_D_ * 4);
  float* tmp    = (float*)alloc((size_t)BK_D_ * 4);
  float* ff     = (float*)alloc((size_t)B_ * K_ * FF_ * 4);
  unsigned short* ffbf = (unsigned short*)alloc((size_t)B_ * K_ * FF_ * 2);

  auto zero = [&](float* p, int n) {
    zero_kernel<<<(n + 255) / 256, 256, 0, stream>>>(p, n);
  };
  auto convT = [&](const float* in, unsigned short* out, int Nd, int Kd) {
    int n = Nd * Kd;
    convT_bf16_kernel<<<(n + 255) / 256, 256, 0, stream>>>(in, out, Nd, Kd);
  };
  auto gemm = [&](const unsigned short* A, const unsigned short* Bt, const float* bias,
                  float* C, int M, int Nd, int Kd, int act) {
    gemm_bf16_kernel<<<dim3(Nd / 128, M / 128), 256, 16384, stream>>>(A, Bt, bias, C, Nd, Kd, act);
  };

  // ---- weight conversion (f32 -> bf16, transposed to [N][K]) -------------
  for (int l = 0; l < L_GNN_; ++l) convT(gnn_W + l * DxD, gnnWb + l * DxD, DD_, DD_);
  for (int l = 0; l < L_TR_; ++l) {
    convT(Wq + l * DxD, Wqb + l * DxD, DD_, DD_);
    convT(Wk + l * DxD, Wkb + l * DxD, DD_, DD_);
    convT(Wv + l * DxD, Wvb + l * DxD, DD_, DD_);
    convT(Wo + l * DxD, Wob + l * DxD, DD_, DD_);
    convT(W1 + l * (size_t)DD_ * FF_, W1b + l * (size_t)DD_ * FF_, FF_, DD_);
    convT(W2 + l * (size_t)FF_ * DD_, W2b + l * (size_t)FF_ * DD_, DD_, FF_);
  }

  // ---- GNN over 32 slots x 2 proteins ------------------------------------
  zero(pool, K_ * B_ * DD_);
  const int ND = NN_ * DD_;                 // 1572864
  const int ndBlocks = ND / 256;            // 6144
  const int edBlocks = (EE_ * DD_) / 256;   // 49152
  for (int k = 0; k < K_; ++k) {
    for (int p = 0; p < 2; ++p) {
      const int* ids = (p == 0 ? x0 : x1) + k * NN_;
      const int* eg  = (p == 0 ? edge0 : edge1) + (size_t)k * 2 * EE_;
      const int* bat = (p == 0 ? batch0 : batch1) + k * NN_;
      gather_kernel<<<ndBlocks, 256, 0, stream>>>(amino, ids, hnodeA);
      float* hcur = hnodeA; float* hnext = hnodeB;
      for (int l = 0; l < L_GNN_; ++l) {
        zero(agg, ND);
        scatter_add_kernel<<<edBlocks, 256, 0, stream>>>(hcur, eg, eg + EE_, agg);
        add_conv_kernel<<<ndBlocks, 256, 0, stream>>>(hcur, agg, xbf, ND);
        gemm(xbf, gnnWb + l * DxD, gnn_b + l * DD_, hnext, NN_, DD_, DD_, /*relu*/1);
        float* t = hcur; hcur = hnext; hnext = t;
      }
      zero(sums, B_ * DD_);
      zero(cnt, B_);
      readout_kernel<<<ndBlocks, 256, 0, stream>>>(hcur, bat, sums, cnt);
      pool_accum_kernel<<<(B_ * DD_ + 255) / 256, 256, 0, stream>>>(
          sums, cnt, pool + (size_t)k * B_ * DD_);
    }
  }

  // ---- embeddings + LN ----------------------------------------------------
  build_feats_kernel<<<(BK_D_ + 255) / 256, 256, 0, stream>>>(
      raw, pool, role_ids, pos_ids, hop_ids, role_emb, pos_emb, hop_emb, hA);
  ln_kernel<<<B_ * K_, 256, 0, stream>>>(hA, nullptr, eln_s, eln_b, hA);

  // ---- transformer layers -------------------------------------------------
  const int M = B_ * K_;                    // 1024 rows
  const int cBlocks = (BK_D_ + 255) / 256;
  const int fBlocks = (B_ * K_ * FF_ + 255) / 256;
  for (int l = 0; l < L_TR_; ++l) {
    conv_bf16_kernel<<<cBlocks, 256, 0, stream>>>(hA, hbf, BK_D_);
    gemm(hbf, Wqb + l * DxD, bq + l * DD_, qb,  M, DD_, DD_, 0);
    gemm(hbf, Wkb + l * DxD, bk + l * DD_, kb2, M, DD_, DD_, 0);
    gemm(hbf, Wvb + l * DxD, bv + l * DD_, vb2, M, DD_, DD_, 0);
    attn_kernel<<<B_ * HH_, 256, 0, stream>>>(qb, kb2, vb2, ctx);
    conv_bf16_kernel<<<cBlocks, 256, 0, stream>>>(ctx, hbf, BK_D_);
    gemm(hbf, Wob + l * DxD, bo + l * DD_, tmp, M, DD_, DD_, 0);
    ln_kernel<<<M, 256, 0, stream>>>(tmp, hA, ln1s + l * DD_, ln1b + l * DD_, hB);
    conv_bf16_kernel<<<cBlocks, 256, 0, stream>>>(hB, hbf, BK_D_);
    gemm(hbf, W1b + l * (size_t)DD_ * FF_, b1 + l * FF_, ff, M, FF_, DD_, /*gelu*/2);
    conv_bf16_kernel<<<fBlocks, 256, 0, stream>>>(ff, ffbf, B_ * K_ * FF_);
    gemm(ffbf, W2b + l * (size_t)FF_ * DD_, b2 + l * DD_, tmp, M, DD_, FF_, 0);
    float* outdst = (l == L_TR_ - 1) ? outp : hA;
    ln_kernel<<<M, 256, 0, stream>>>(tmp, hB, ln2s + l * DD_, ln2b + l * DD_, outdst);
  }
}
